// LSTMDipNetDecoder_55078660604162
// MI455X (gfx1250) — matrix operand order repaired and checked by reading, not compiled, not run on hardware
//
#include <hip/hip_runtime.h>
#include <cstdint>
#include <cstddef>

// ---- problem constants ----
#define BB   512
#define SS   17
#define LL   81
#define VV   13030
#define ENCD 240
#define OED  80
#define PED  60
#define HD   200
#define IND  380          // ENC + OE + PE
#define G4H  800          // 4*H
#define GP   832          // G4H padded to 13 groups of 64 (branch-free GEMM)
#define KXH  608          // padded K for [x|h]: 380+200=580 -> 19*32
#define KH   224          // padded K for h: 200 -> 7*32
#define VP   13056        // VV padded to 204 groups of 64
#define SV   (SS*VV)

typedef __attribute__((ext_vector_type(16))) __bf16 v16bf;
typedef __attribute__((ext_vector_type(8)))  float  v8f;
typedef __attribute__((ext_vector_type(4)))  unsigned int u32x4;

union Frag { v16bf v; u32x4 u[2]; };

__device__ __forceinline__ unsigned short f2bf(float f) {
    unsigned int x = __float_as_uint(f);
    unsigned int r = x + 0x7fffu + ((x >> 16) & 1u);   // round-to-nearest-even
    return (unsigned short)(r >> 16);
}

__device__ __forceinline__ float sigm(float x) { return 1.0f / (1.0f + __expf(-x)); }

__device__ __forceinline__ unsigned int mix3(unsigned int a, unsigned int b, unsigned int c) {
    unsigned int h = a * 0x9E3779B1u + b * 0x85EBCA77u + c * 0xC2B2AE3Du + 0x27220A95u;
    h ^= h >> 15; h *= 0x2C1B3C6Du;
    h ^= h >> 12; h *= 0x297A2D39u;
    h ^= h >> 15;
    return h;
}

// ---------------- one-time prep kernels ----------------

// Wcat[832][608] bf16 = [W_ih (380) | W_hh (200) | K-pad] rows; rows >= 800 are zero
__global__ void conv_wcat(const float* __restrict__ Wih, const float* __restrict__ Whh,
                          unsigned short* __restrict__ Wcat) {
    int tid = blockIdx.x * blockDim.x + threadIdx.x;
    if (tid >= GP * KXH) return;
    int n = tid / KXH, k = tid % KXH;
    float v = 0.0f;
    if (n < G4H) {
        if (k < IND)            v = Wih[(size_t)n * IND + k];
        else if (k < IND + HD)  v = Whh[(size_t)n * HD + (k - IND)];
    }
    Wcat[tid] = f2bf(v);
}

// Wout[13056][224] bf16 = out_W zero-padded in K and N
__global__ void conv_wout(const float* __restrict__ W, unsigned short* __restrict__ Wb) {
    int tid = blockIdx.x * blockDim.x + threadIdx.x;
    if (tid >= VP * KH) return;
    int n = tid / KH, k = tid % KH;
    Wb[tid] = f2bf((n < VV && k < HD) ? W[(size_t)n * HD + k] : 0.0f);
}

// power_emb[B][60] = power_1h @ power_W^T + power_b
__global__ void power_emb_k(const float* __restrict__ p1h, const float* __restrict__ pW,
                            const float* __restrict__ pb, float* __restrict__ pe) {
    int tid = blockIdx.x * blockDim.x + threadIdx.x;
    if (tid >= BB * PED) return;
    int b = tid / PED, p = tid % PED;
    float v = pb[p];
    #pragma unroll
    for (int j = 0; j < 7; ++j) v += p1h[b * 7 + j] * pW[p * 7 + j];
    pe[tid] = v;
}

__global__ void invalid_k(const int* __restrict__ loc, int* __restrict__ inv) {
    int b = blockIdx.x * blockDim.x + threadIdx.x;
    if (b >= BB) return;
    bool any = false;
    for (int l = 0; l < LL; ++l) any |= (loc[b * LL + l] != -1);
    inv[b] = any ? 0 : 1;
}

__global__ void init_k(float* __restrict__ c, unsigned short* __restrict__ hbf,
                       unsigned short* __restrict__ oebf) {
    int tid = blockIdx.x * blockDim.x + threadIdx.x;
    if (tid < BB * KH)  hbf[tid] = 0;
    if (tid < BB * HD)  c[tid]   = 0.0f;
    if (tid < BB * OED) oebf[tid] = 0;
}

// ---------------- per-step kernels ----------------

// alignment + loc_enc + assemble xh row [loc_enc|order_emb|power_emb|h|pad] in bf16
__global__ void prestep(const int* __restrict__ loc_idxs, const float* __restrict__ MA,
                        const float* __restrict__ enc, const unsigned short* __restrict__ oe_bf,
                        const float* __restrict__ pe, const unsigned short* __restrict__ hbf,
                        unsigned short* __restrict__ xh, int step) {
    __shared__ float al[LL];
    __shared__ int   li[LL];
    int b = blockIdx.x, t = threadIdx.x;
    if (t < LL) li[t] = loc_idxs[b * LL + t];
    __syncthreads();
    if (t < LL) {
        float a = 0.0f;
        for (int l = 0; l < LL; ++l) {
            int v = li[l];
            if (v == step || v == -2) a += MA[l * LL + t];
        }
        al[t] = a;
    }
    __syncthreads();
    float denom = 0.0f;
    for (int l = 0; l < LL; ++l) denom += al[l];
    float inv = (denom != 0.0f) ? 1.0f / denom : 0.0f;

    unsigned short* xrow = xh + (size_t)b * KXH;
    for (int d = t; d < ENCD; d += blockDim.x) {
        float v = 0.0f;
        const float* ep = enc + ((size_t)b * LL) * ENCD + d;
        for (int l = 0; l < LL; ++l) v += al[l] * ep[(size_t)l * ENCD];
        xrow[d] = f2bf(v * inv);
    }
    for (int k = t; k < OED; k += blockDim.x) xrow[ENCD + k] = oe_bf[b * OED + k];
    for (int k = t; k < PED; k += blockDim.x) xrow[ENCD + OED + k] = f2bf(pe[b * PED + k]);
    for (int k = t; k < HD;  k += blockDim.x) xrow[IND + k] = hbf[b * KH + k];
    for (int k = IND + HD + t; k < KXH; k += blockDim.x) xrow[k] = 0;
}

// gates[512][832] = xh[512][608] @ Wcat[832][608]^T  (bf16 WMMA, f32 acc)
// Grid is exact (52 blocks * 8 waves == 32*13 jobs): fully branch-free.
__global__ void gates_gemm(const unsigned short* __restrict__ xh,
                           const unsigned short* __restrict__ Wcat,
                           float* __restrict__ gates) {
    const int MT = BB / 16;                 // 32
    int wave = blockIdx.x * (blockDim.x >> 5) + (threadIdx.x >> 5);
    int mt = wave % MT, ng = wave / MT;     // ng in [0,13)
    int m0 = mt * 16, n0 = ng * 64;
    int lane = threadIdx.x & 31, lr = lane & 15, lh = lane >> 4;

    v8f acc[4] = {};
    for (int k0 = 0; k0 < KXH; k0 += 32) {
        Frag a;
        const unsigned short* ap = xh + (size_t)(m0 + lr) * KXH + k0 + lh * 8;
        a.u[0] = *(const u32x4*)ap;
        a.u[1] = *(const u32x4*)(ap + 16);
        #pragma unroll
        for (int tt = 0; tt < 4; ++tt) {
            Frag bfr;
            const unsigned short* bp = Wcat + (size_t)(n0 + tt * 16 + lr) * KXH + k0 + lh * 8;
            bfr.u[0] = *(const u32x4*)bp;
            bfr.u[1] = *(const u32x4*)(bp + 16);
            acc[tt] = __builtin_amdgcn_wmma_f32_16x16x32_bf16(
                false, a.v, false, bfr.v, (short)0, acc[tt], false, false);
        }
    }
    #pragma unroll
    for (int tt = 0; tt < 4; ++tt) {
        int n = n0 + tt * 16 + lr;
        #pragma unroll
        for (int r = 0; r < 8; ++r) {
            int m = m0 + r + lh * 8;
            gates[(size_t)m * GP + n] = acc[tt][r];
        }
    }
}

// LSTM cell: consumes gates (stride GP) + biases, updates c, writes padded bf16 h
__global__ void lstm_cell(const float* __restrict__ gates, const float* __restrict__ b_ih,
                          const float* __restrict__ b_hh, float* __restrict__ c,
                          unsigned short* __restrict__ hbf) {
    int tid = blockIdx.x * blockDim.x + threadIdx.x;
    if (tid >= BB * HD) return;
    int b = tid / HD, j = tid % HD;
    const float* g = gates + (size_t)b * GP;
    float ig = g[j]          + b_ih[j]          + b_hh[j];
    float fg = g[HD + j]     + b_ih[HD + j]     + b_hh[HD + j];
    float gg = g[2 * HD + j] + b_ih[2 * HD + j] + b_hh[2 * HD + j];
    float og = g[3 * HD + j] + b_ih[3 * HD + j] + b_hh[3 * HD + j];
    float cn = sigm(fg) * c[tid] + sigm(ig) * tanhf(gg);
    float hn = sigm(og) * tanhf(cn);
    c[tid] = cn;
    hbf[(size_t)b * KH + j] = f2bf(hn);
}

// scores(b,v) = h @ out_W^T + out_b, then masked min(), streamed to d_out.
// Grid is exact (816 blocks * 8 waves == 32*204 jobs). Loads branch-free
// (Wout padded to 13056 rows); only the epilogue store is guarded.
__global__ void scores_gemm(const unsigned short* __restrict__ hbf,
                            const unsigned short* __restrict__ Wout,
                            const float* __restrict__ out_b,
                            const int* __restrict__ maskBase,   // order_masks + s*V
                            const int* __restrict__ invalid,
                            float* __restrict__ scoresBase) {   // d_out + B*S + s*V
    const int MT = BB / 16;                  // 32
    int wave = blockIdx.x * (blockDim.x >> 5) + (threadIdx.x >> 5);
    int mt = wave % MT, ng = wave / MT;      // ng in [0,204)
    int m0 = mt * 16, n0 = ng * 64;
    int lane = threadIdx.x & 31, lr = lane & 15, lh = lane >> 4;

    v8f acc[4] = {};
    for (int k0 = 0; k0 < KH; k0 += 32) {
        Frag a;
        const unsigned short* ap = hbf + (size_t)(m0 + lr) * KH + k0 + lh * 8;
        a.u[0] = *(const u32x4*)ap;
        a.u[1] = *(const u32x4*)(ap + 16);
        #pragma unroll
        for (int tt = 0; tt < 4; ++tt) {
            Frag bfr;
            const unsigned short* bp = Wout + (size_t)(n0 + tt * 16 + lr) * KH + k0 + lh * 8;
            bfr.u[0] = *(const u32x4*)bp;
            bfr.u[1] = *(const u32x4*)(bp + 16);
            acc[tt] = __builtin_amdgcn_wmma_f32_16x16x32_bf16(
                false, a.v, false, bfr.v, (short)0, acc[tt], false, false);
        }
    }
    #pragma unroll
    for (int tt = 0; tt < 4; ++tt) {
        int col = n0 + tt * 16 + lr;
        if (col < VV) {                       // epilogue-only divergence
            float bias = out_b[col];
            #pragma unroll
            for (int r = 0; r < 8; ++r) {
                int brow = m0 + r + lh * 8;   // batch index
                float v = acc[tt][r] + bias;
                bool mb = (invalid[brow] != 0) || (maskBase[(size_t)brow * SV + col] != 0);
                float lim = mb ? 9.0e8f : -1.0e8f;
                scoresBase[(size_t)brow * SV + col] = fminf(v, lim);
            }
        }
    }
}

// Gumbel-max sample over masked scores; write idx (as float) and gather embedding
__global__ void sample_k(const float* __restrict__ scoresBase,  // d_out + B*S + s*V
                         const float* __restrict__ order_embedding,
                         unsigned short* __restrict__ oe_bf,
                         float* __restrict__ out_idx, int step) {
    __shared__ float sv[256];
    __shared__ int   si[256];
    int b = blockIdx.x, t = threadIdx.x;
    const float* row = scoresBase + (size_t)b * SV;
    float best = -3.4e38f; int bi = 0;
    for (int v = t; v < VV; v += 256) {
        unsigned int h = mix3((unsigned int)v, (unsigned int)b, (unsigned int)step);
        float u = ((float)h + 0.5f) * (1.0f / 4294967296.0f);
        float g = -__logf(-__logf(u));
        float key = row[v] + g;               // TEMPERATURE == 1
        if (key > best) { best = key; bi = v; }
    }
    sv[t] = best; si[t] = bi;
    __syncthreads();
    for (int off = 128; off > 0; off >>= 1) {
        if (t < off && sv[t + off] > sv[t]) { sv[t] = sv[t + off]; si[t] = si[t + off]; }
        __syncthreads();
    }
    int widx = si[0];
    if (t == 0) out_idx[b * SS + step] = (float)widx;
    if (t < OED) oe_bf[b * OED + t] = f2bf(order_embedding[(size_t)widx * OED + t]);
}

// ---------------- host launcher ----------------

static inline size_t rup(size_t x) { return (x + 255) & ~(size_t)255; }

extern "C" void kernel_launch(void* const* d_in, const int* in_sizes, int n_in,
                              void* d_out, int out_size, void* d_ws, size_t ws_size,
                              hipStream_t stream) {
    (void)in_sizes; (void)n_in; (void)out_size; (void)ws_size;
    const float* enc      = (const float*)d_in[0];
    const int*   loc_idxs = (const int*)d_in[2];
    const int*   masks    = (const int*)d_in[3];
    const float* p1h      = (const float*)d_in[4];
    const float* oemb     = (const float*)d_in[5];
    const float* pW       = (const float*)d_in[6];
    const float* pb       = (const float*)d_in[7];
    const float* Wih      = (const float*)d_in[8];
    const float* Whh      = (const float*)d_in[9];
    const float* b_ih     = (const float*)d_in[10];
    const float* b_hh     = (const float*)d_in[11];
    const float* outW     = (const float*)d_in[12];
    const float* outb     = (const float*)d_in[13];
    const float* MA       = (const float*)d_in[14];

    char* ws = (char*)d_ws;
    size_t off = 0;
    unsigned short* wcat = (unsigned short*)(ws + off); off = rup(off + (size_t)GP * KXH * 2);
    unsigned short* wout = (unsigned short*)(ws + off); off = rup(off + (size_t)VP * KH * 2);
    float*          pe   = (float*)(ws + off);          off = rup(off + (size_t)BB * PED * 4);
    int*            inv  = (int*)(ws + off);            off = rup(off + (size_t)BB * 4);
    unsigned short* xh   = (unsigned short*)(ws + off); off = rup(off + (size_t)BB * KXH * 2);
    float*          gts  = (float*)(ws + off);          off = rup(off + (size_t)BB * GP * 4);
    float*          cst  = (float*)(ws + off);          off = rup(off + (size_t)BB * HD * 4);
    unsigned short* hbf  = (unsigned short*)(ws + off); off = rup(off + (size_t)BB * KH * 2);
    unsigned short* oebf = (unsigned short*)(ws + off); off = rup(off + (size_t)BB * OED * 2);

    float* out_idx    = (float*)d_out;             // [B, S] indices (as float)
    float* out_scores = (float*)d_out + BB * SS;   // [B, S, V]

    // one-time prep (cheap vs. 453MB score writes)
    conv_wcat<<<(GP * KXH + 255) / 256, 256, 0, stream>>>(Wih, Whh, wcat);
    conv_wout<<<(VP * KH + 255) / 256, 256, 0, stream>>>(outW, wout);
    power_emb_k<<<(BB * PED + 255) / 256, 256, 0, stream>>>(p1h, pW, pb, pe);
    invalid_k<<<(BB + 255) / 256, 256, 0, stream>>>(loc_idxs, inv);
    init_k<<<(BB * KH + 255) / 256, 256, 0, stream>>>(cst, hbf, oebf);

    const int gatesBlocks = (32 * 13) / 8;     // 52 blocks, exact
    const int scoreBlocks = (32 * 204) / 8;    // 816 blocks, exact

    for (int s = 0; s < SS; ++s) {
        prestep<<<BB, 128, 0, stream>>>(loc_idxs, MA, enc, oebf, pe, hbf, xh, s);
        gates_gemm<<<gatesBlocks, 256, 0, stream>>>(xh, wcat, gts);
        lstm_cell<<<(BB * HD + 255) / 256, 256, 0, stream>>>(gts, b_ih, b_hh, cst, hbf);
        scores_gemm<<<scoreBlocks, 256, 0, stream>>>(
            hbf, wout, outb, masks + (size_t)s * VV, inv,
            out_scores + (size_t)s * VV);
        sample_k<<<BB, 256, 0, stream>>>(out_scores + (size_t)s * VV, oemb, oebf, out_idx, s);
    }
}
